// DecoderRNN_41317585387632
// MI455X (gfx1250) — matrix operand order repaired and checked by reading, not compiled, not run on hardware
//
#include <hip/hip_runtime.h>
#include <hip/hip_bf16.h>

typedef __bf16 bf16_t;
typedef __attribute__((ext_vector_type(16))) __bf16 v16bf;
typedef __attribute__((ext_vector_type(8)))  float  v8f;

#define Bsz 256
#define Hsz 1024
#define Lsz 512
#define Vsz 32000

// ---------------------------------------------------------------------------
// Embedding gather + concat: embh = bf16([emb[input], h])  [B, 2H]
// also x2[:, 0:H] = bf16(embedded)
// ---------------------------------------------------------------------------
__global__ void k_embed(const int* __restrict__ input,
                        const float* __restrict__ hidden,
                        const float* __restrict__ emb,
                        bf16_t* __restrict__ embh,
                        bf16_t* __restrict__ x2) {
  int b = blockIdx.x;
  int row = input[b];
  for (int c = threadIdx.x; c < 2 * Hsz; c += blockDim.x) {
    bf16_t v;
    if (c < Hsz) {
      v = (bf16_t)emb[(size_t)row * Hsz + c];
      x2[(size_t)b * 2 * Hsz + c] = v;
    } else {
      v = (bf16_t)hidden[(size_t)b * Hsz + (c - Hsz)];
    }
    embh[(size_t)b * 2 * Hsz + c] = v;
  }
}

// ---------------------------------------------------------------------------
// Generic bf16 WMMA GEMM.
//  TRANSW=1 : Y[m,n] = sum_k A[m,k] * W[n,k]   (PyTorch Linear, W [N,K])
//  TRANSW=0 : Y[m,n] = sum_k A[m,k] * W[k,n]   (plain A@B, W [K,N])
// A is bf16 row-major (lda), W fp32 (converted to bf16 in-register).
// One wave per 16x16 C tile; 4 waves/block share the same W tile (same tn).
// ---------------------------------------------------------------------------
template<bool TRANSW, bool RELU, bool OUTBF16>
__global__ void k_gemm(const bf16_t* __restrict__ A, int lda,
                       const float*  __restrict__ W, int ldw,
                       const float*  __restrict__ bias,
                       void* __restrict__ C, int ldc,
                       int M, int N, int K) {
  const int lane = threadIdx.x & 31;
  const int wave = blockIdx.x * (blockDim.x >> 5) + (threadIdx.x >> 5);
  const int tilesM = M >> 4;
  const int tm = wave % tilesM;
  const int tn = wave / tilesM;
  if (tn * 16 >= N) return;           // wave-uniform: EXEC stays all-ones

  const int hi = lane >> 4;           // lane group (0: lanes 0-15, 1: 16-31)
  const int lm = lane & 15;
  const int m0 = tm * 16, n0 = tn * 16;
  const bf16_t* __restrict__ arow = A + (size_t)(m0 + lm) * lda;

  v8f acc = {};
  if (TRANSW) {
    const float* __restrict__ wrow = W + (size_t)(n0 + lm) * ldw;
    for (int k0 = 0; k0 < K; k0 += 32) {
      if (k0 + 128 < K) __builtin_prefetch(wrow + k0 + 128, 0, 0);
      v16bf a, b;
#pragma unroll
      for (int j = 0; j < 8; ++j) {
        const int ka = 2 * j + ((j >= 4) ? 8 : 0) + 8 * hi + k0; // ISA A 16x32 layout
        a[2 * j]     = arow[ka];
        a[2 * j + 1] = arow[ka + 1];
        const int kb = 2 * j + ((j >= 4) ? 8 : 0) + 8 * hi + k0; // ISA B 32x16 layout
        b[2 * j]     = (bf16_t)wrow[kb];
        b[2 * j + 1] = (bf16_t)wrow[kb + 1];
      }
      acc = __builtin_amdgcn_wmma_f32_16x16x32_bf16(false, a, false, b,
                                                    (short)0, acc, false, false);
    }
  } else {
    for (int k0 = 0; k0 < K; k0 += 32) {
      v16bf a, b;
#pragma unroll
      for (int j = 0; j < 8; ++j) {
        const int ka = 2 * j + ((j >= 4) ? 8 : 0) + 8 * hi + k0;
        a[2 * j]     = arow[ka];
        a[2 * j + 1] = arow[ka + 1];
        const int kb = 2 * j + ((j >= 4) ? 8 : 0) + 8 * hi + k0;
        b[2 * j]     = (bf16_t)W[(size_t)kb * ldw + n0 + lm];
        b[2 * j + 1] = (bf16_t)W[(size_t)(kb + 1) * ldw + n0 + lm];
      }
      acc = __builtin_amdgcn_wmma_f32_16x16x32_bf16(false, a, false, b,
                                                    (short)0, acc, false, false);
    }
  }

  const float bv = bias ? bias[n0 + lm] : 0.0f;
#pragma unroll
  for (int r = 0; r < 8; ++r) {
    const int m = m0 + r + 8 * hi;    // ISA C/D 16x16 f32 layout
    float v = acc[r] + bv;
    if (RELU) v = fmaxf(v, 0.0f);
    if (OUTBF16) ((bf16_t*)C)[(size_t)m * ldc + n0 + lm] = (bf16_t)v;
    else         ((float*) C)[(size_t)m * ldc + n0 + lm] = v;
  }
}

// ---------------------------------------------------------------------------
// Row softmax over L=512: writes fp32 to d_out and bf16 copy for next GEMM A.
// ---------------------------------------------------------------------------
__global__ void k_softmax(const float* __restrict__ logits,
                          float* __restrict__ out_w,
                          bf16_t* __restrict__ awbf) {
  __shared__ float red[256];
  const int b = blockIdx.x, t = threadIdx.x;
  const float* row = logits + (size_t)b * Lsz;
  float m = fmaxf(row[t], row[t + 256]);
  red[t] = m; __syncthreads();
  for (int s = 128; s > 0; s >>= 1) {
    if (t < s) red[t] = fmaxf(red[t], red[t + s]);
    __syncthreads();
  }
  const float mx = red[0]; __syncthreads();
  const float e0 = __expf(row[t] - mx), e1 = __expf(row[t + 256] - mx);
  red[t] = e0 + e1; __syncthreads();
  for (int s = 128; s > 0; s >>= 1) {
    if (t < s) red[t] += red[t + s];
    __syncthreads();
  }
  const float inv = 1.0f / red[0];
  const float w0 = e0 * inv, w1 = e1 * inv;
  out_w[(size_t)b * Lsz + t]       = w0;
  out_w[(size_t)b * Lsz + t + 256] = w1;
  awbf[(size_t)b * Lsz + t]        = (bf16_t)w0;
  awbf[(size_t)b * Lsz + t + 256]  = (bf16_t)w1;
}

// ---------------------------------------------------------------------------
// GRU pointwise (PyTorch gate order r,z,n): h_new fp32 to d_out + bf16 copy.
// ---------------------------------------------------------------------------
__global__ void k_gru(const float* __restrict__ gi, const float* __restrict__ gh,
                      const float* __restrict__ hidden,
                      float* __restrict__ h_out, bf16_t* __restrict__ hnbf) {
  const int idx = blockIdx.x * blockDim.x + threadIdx.x;
  if (idx >= Bsz * Hsz) return;
  const int b = idx >> 10, j = idx & (Hsz - 1);
  const float* gib = gi + (size_t)b * 3 * Hsz;
  const float* ghb = gh + (size_t)b * 3 * Hsz;
  const float r = 1.0f / (1.0f + __expf(-(gib[j] + ghb[j])));
  const float z = 1.0f / (1.0f + __expf(-(gib[Hsz + j] + ghb[Hsz + j])));
  const float n = tanhf(gib[2 * Hsz + j] + r * ghb[2 * Hsz + j]);
  const float h = hidden[idx];
  const float hn = (1.0f - z) * n + z * h;
  h_out[idx] = hn;
  hnbf[idx]  = (bf16_t)hn;
}

// ---------------------------------------------------------------------------
// In-place log-softmax over V=32000 per row.
// ---------------------------------------------------------------------------
__global__ void k_logsoftmax(float* __restrict__ logits) {
  __shared__ float red[512];
  const int b = blockIdx.x, t = threadIdx.x;
  float* row = logits + (size_t)b * Vsz;
  float m = -INFINITY;
  for (int i = t; i < Vsz; i += 512) m = fmaxf(m, row[i]);
  red[t] = m; __syncthreads();
  for (int s = 256; s > 0; s >>= 1) {
    if (t < s) red[t] = fmaxf(red[t], red[t + s]);
    __syncthreads();
  }
  const float mx = red[0]; __syncthreads();
  float sum = 0.0f;
  for (int i = t; i < Vsz; i += 512) sum += __expf(row[i] - mx);
  red[t] = sum; __syncthreads();
  for (int s = 256; s > 0; s >>= 1) {
    if (t < s) red[t] += red[t + s];
    __syncthreads();
  }
  const float lse = mx + __logf(red[0]);
  for (int i = t; i < Vsz; i += 512) row[i] = row[i] - lse;
}

// ---------------------------------------------------------------------------
extern "C" void kernel_launch(void* const* d_in, const int* in_sizes, int n_in,
                              void* d_out, int out_size, void* d_ws, size_t ws_size,
                              hipStream_t stream) {
  const int*   input   = (const int*)  d_in[0];
  const float* hidden  = (const float*)d_in[1];
  const float* enc     = (const float*)d_in[2];
  const float* emb     = (const float*)d_in[3];
  const float* attn_W  = (const float*)d_in[4];
  const float* attn_b  = (const float*)d_in[5];
  const float* comb_W  = (const float*)d_in[6];
  const float* comb_b  = (const float*)d_in[7];
  const float* W_ih    = (const float*)d_in[8];
  const float* W_hh    = (const float*)d_in[9];
  const float* b_ih    = (const float*)d_in[10];
  const float* b_hh    = (const float*)d_in[11];
  const float* out_W   = (const float*)d_in[12];
  const float* out_b   = (const float*)d_in[13];

  // d_out: [output B*V | h_new B*H | attn_weights B*L]
  float* out_logits = (float*)d_out;
  float* out_h      = out_logits + (size_t)Bsz * Vsz;
  float* out_attn   = out_h + (size_t)Bsz * Hsz;

  // workspace carve-up (~9.8 MB)
  char* w = (char*)d_ws;
  bf16_t* embh = (bf16_t*)w; w += (size_t)Bsz * 2 * Hsz * 2;   // [B,2H] bf16 (emb|h)
  bf16_t* x2   = (bf16_t*)w; w += (size_t)Bsz * 2 * Hsz * 2;   // [B,2H] bf16 (emb|applied)
  float*  alog = (float*) w; w += (size_t)Bsz * Lsz * 4;       // attn logits fp32
  bf16_t* awbf = (bf16_t*)w; w += (size_t)Bsz * Lsz * 2;       // attn weights bf16
  bf16_t* xbf  = (bf16_t*)w; w += (size_t)Bsz * Hsz * 2;       // relu(comb) bf16
  float*  gi   = (float*) w; w += (size_t)Bsz * 3 * Hsz * 4;   // [B,3H]
  float*  gh   = (float*) w; w += (size_t)Bsz * 3 * Hsz * 4;   // [B,3H]
  bf16_t* hnbf = (bf16_t*)w; w += (size_t)Bsz * Hsz * 2;       // h_new bf16

  // 1. gather + concat
  k_embed<<<Bsz, 256, 0, stream>>>(input, hidden, emb, embh, x2);

  // 2. attn_logits = [emb,h] @ attn_W^T + attn_b   (M=256,N=512,K=2048)
  k_gemm<true, false, false><<<128, 128, 0, stream>>>(
      embh, 2 * Hsz, attn_W, 2 * Hsz, attn_b, alog, Lsz, Bsz, Lsz, 2 * Hsz);

  // 3. softmax over L
  k_softmax<<<Bsz, 256, 0, stream>>>(alog, out_attn, awbf);

  // 4. attn_applied = aw @ enc   (M=256,N=1024,K=512), bf16 into x2[:,H:2H]
  k_gemm<false, false, true><<<256, 128, 0, stream>>>(
      awbf, Lsz, enc, Hsz, nullptr, x2 + Hsz, 2 * Hsz, Bsz, Hsz, Lsz);

  // 5. x = relu([emb,applied] @ comb_W^T + comb_b)  (M=256,N=1024,K=2048)
  k_gemm<true, true, true><<<256, 128, 0, stream>>>(
      x2, 2 * Hsz, comb_W, 2 * Hsz, comb_b, xbf, Hsz, Bsz, Hsz, 2 * Hsz);

  // 6. gi = x @ W_ih^T + b_ih   (M=256,N=3072,K=1024)
  k_gemm<true, false, false><<<768, 128, 0, stream>>>(
      xbf, Hsz, W_ih, Hsz, b_ih, gi, 3 * Hsz, Bsz, 3 * Hsz, Hsz);

  // 7. gh = h @ W_hh^T + b_hh   (A = embh[:, H:2H], lda=2H)
  k_gemm<true, false, false><<<768, 128, 0, stream>>>(
      embh + Hsz, 2 * Hsz, W_hh, Hsz, b_hh, gh, 3 * Hsz, Bsz, 3 * Hsz, Hsz);

  // 8. GRU pointwise -> h_new
  k_gru<<<(Bsz * Hsz + 255) / 256, 256, 0, stream>>>(gi, gh, hidden, out_h, hnbf);

  // 9. logits = h_new @ out_W^T + out_b   (M=256,N=32000,K=1024)
  k_gemm<true, false, false><<<8000, 128, 0, stream>>>(
      hnbf, Hsz, out_W, Hsz, out_b, out_logits, Vsz, Bsz, Vsz, Hsz);

  // 10. in-place log-softmax over V
  k_logsoftmax<<<Bsz, 512, 0, stream>>>(out_logits);
}
